// myGAT_25494925869744
// MI455X (gfx1250) — matrix operand order repaired and checked by reading, not compiled, not run on hardware
//
#include <hip/hip_runtime.h>

// ---- model dims ----
#define NN     20000
#define EE_    320000
#define EDIMC  64
#define NETC   8
#define ALPHAC 0.05f
#define NEGC   0.2f

typedef _Float16 v16h __attribute__((ext_vector_type(16)));
typedef _Float16 h4   __attribute__((ext_vector_type(4)));
typedef float    v8f  __attribute__((ext_vector_type(8)));

// ---------------------------------------------------------------------------
// WMMA GEMM: C[M,Nout] = A[M,K] @ W[K,Nout] (+bias) (+=C if accumulate)
// f32 in memory, f16 staged in LDS (double-buffered), f32 accumulation via
// v_wmma_f32_16x16x32_f16. 256 threads = 8 waves; wave -> 16x16 C tile,
// block -> 128x16. K is compile-time (fully unrolled, software pipelined).
// ---------------------------------------------------------------------------
template<int K>
__global__ __launch_bounds__(256)
void gemm_wmma(const float* __restrict__ A, const float* __restrict__ W,
               const float* __restrict__ bias, float* __restrict__ C,
               int M, int Nout, int accumulate)
{
    constexpr int KT = K / 32;
    __shared__ _Float16 As[2][128][40];   // 128 rows x 32 K (padded)
    __shared__ _Float16 Bt[2][16][40];    // B transposed: [n][k]

    const int tid  = threadIdx.x;
    const int lane = tid & 31;
    const int wave = tid >> 5;
    const int g    = lane >> 4;
    const int lm   = lane & 15;
    const int rowBase = blockIdx.y * 128;
    const int colBase = blockIdx.x * 16;

    // A staging: thread -> rows (tid>>3)+32*i, cols (tid&7)*4 .. +3  (float4)
    const int arow0 = tid >> 3;
    const int acol  = (tid & 7) * 4;
    // B staging: thread -> kk=(tid>>4)+16*i, nn=tid&15
    const int bk0 = tid >> 4;
    const int bn  = tid & 15;

    float4 va[4];
    float  wb[2];

    auto load_regs = [&](int t) {
        const int k0 = t * 32;
        #pragma unroll
        for (int i = 0; i < 4; ++i) {
            int gr = rowBase + arow0 + 32 * i;
            gr = (gr < M) ? gr : (M - 1);   // clamp: rows >= M never stored
            va[i] = *(const float4*)&A[(size_t)gr * K + k0 + acol];
        }
        #pragma unroll
        for (int i = 0; i < 2; ++i)
            wb[i] = W[(size_t)(k0 + bk0 + 16 * i) * Nout + colBase + bn];
    };
    auto store_lds = [&](int p) {
        #pragma unroll
        for (int i = 0; i < 4; ++i) {
            h4 hv;
            hv[0] = (_Float16)va[i].x; hv[1] = (_Float16)va[i].y;
            hv[2] = (_Float16)va[i].z; hv[3] = (_Float16)va[i].w;
            *(h4*)&As[p][arow0 + 32 * i][acol] = hv;
        }
        #pragma unroll
        for (int i = 0; i < 2; ++i)
            Bt[p][bn][bk0 + 16 * i] = (_Float16)wb[i];
    };

    v8f acc = {0.f,0.f,0.f,0.f,0.f,0.f,0.f,0.f};

    load_regs(0);
    store_lds(0);
    #pragma unroll
    for (int t = 0; t < KT; ++t) {
        __syncthreads();
        const int p = t & 1;
        if (t + 1 < KT) load_regs(t + 1);   // overlap next-tile loads with WMMA

        union Frag { v16h v; unsigned u[8]; } ua, ub;
        const int arow = wave * 16 + lm;
        #pragma unroll
        for (int vv = 0; vv < 8; ++vv) {
            int ka = ((vv < 4) ? 0 : 16) + g * 8 + 2 * (vv & 3);  // ISA 7.12.2 A layout
            ua.u[vv] = *(const unsigned*)&As[p][arow][ka];
            int kb = ((vv < 4) ? 0 : 16) + g * 8 + 2 * (vv & 3);  // B layout
            ub.u[vv] = *(const unsigned*)&Bt[p][lm][kb];
        }
        acc = __builtin_amdgcn_wmma_f32_16x16x32_f16(
            false, ua.v, false, ub.v, (short)0, acc, false, false);

        if (t + 1 < KT) store_lds(p ^ 1);
    }

    const int n = colBase + lm;
    const float b = bias ? bias[n] : 0.f;
    #pragma unroll
    for (int vv = 0; vv < 8; ++vv) {
        int m = rowBase + wave * 16 + vv + 8 * g;   // C/D: M = v + 8*(lane/16)
        if (m < M) {
            float val = acc[vv] + b;
            size_t o = (size_t)m * Nout + n;
            if (accumulate) val += C[o];
            C[o] = val;
        }
    }
}

// ---------------------------------------------------------------------------
// CSR build (deterministic: per-node edge lists sorted by edge id)
// ---------------------------------------------------------------------------
__global__ void zero_i32(int* p, int n) {
    int gid = blockIdx.x * blockDim.x + threadIdx.x;
    if (gid < n) p[gid] = 0;
}

__global__ void count_kernel(const int* __restrict__ dst, int* counts, int e) {
    int gid = blockIdx.x * blockDim.x + threadIdx.x;
    if (gid < e) atomicAdd(&counts[dst[gid]], 1);
}

__global__ void scan_kernel(const int* __restrict__ counts, int* row_ptr, int n) {
    __shared__ int buf[1024];
    __shared__ int carry;
    if (threadIdx.x == 0) { carry = 0; row_ptr[0] = 0; }
    __syncthreads();
    for (int base = 0; base < n; base += 1024) {
        int i = base + threadIdx.x;
        int v = (i < n) ? counts[i] : 0;
        buf[threadIdx.x] = v;
        __syncthreads();
        for (int off = 1; off < 1024; off <<= 1) {
            int t = (threadIdx.x >= off) ? buf[threadIdx.x - off] : 0;
            __syncthreads();
            buf[threadIdx.x] += t;
            __syncthreads();
        }
        if (i < n) row_ptr[i + 1] = carry + buf[threadIdx.x];
        __syncthreads();
        if (threadIdx.x == 0) carry += buf[1023];
        __syncthreads();
    }
}

__global__ void scatter_kernel(const int* __restrict__ dst, const int* __restrict__ row_ptr,
                               int* cursor, int* eids, int e) {
    int gid = blockIdx.x * blockDim.x + threadIdx.x;
    if (gid < e) {
        int d = dst[gid];
        int p = atomicAdd(&cursor[d], 1);
        eids[row_ptr[d] + p] = gid;
    }
}

__global__ void sort_kernel(const int* __restrict__ row_ptr, int* eids, int n) {
    int i = blockIdx.x * blockDim.x + threadIdx.x;
    if (i >= n) return;
    int s0 = row_ptr[i], s1 = row_ptr[i + 1];
    for (int j = s0 + 1; j < s1; ++j) {
        int key = eids[j];
        int k = j - 1;
        while (k >= s0 && eids[k] > key) { eids[k + 1] = eids[k]; --k; }
        eids[k + 1] = key;
    }
}

// ---------------------------------------------------------------------------
// Per-etype attention bias: ee[t,h] = attn_e[h,:] . (edge_emb[t,:] @ w_e[:, h*EDIM:+EDIM])
// ---------------------------------------------------------------------------
__global__ void ee_kernel(const float* __restrict__ emb, const float* __restrict__ w_e,
                          const float* __restrict__ attn_e, float* ee, int H) {
    int tid = threadIdx.x;
    if (tid >= NETC * H) return;
    int t = tid / H, h = tid % H;
    float acc = 0.f;
    for (int d = 0; d < EDIMC; ++d) {
        float tmp = 0.f;
        for (int j = 0; j < EDIMC; ++j)
            tmp += emb[t * EDIMC + j] * w_e[(size_t)j * H * EDIMC + h * EDIMC + d];
        acc += tmp * attn_e[h * EDIMC + d];
    }
    ee[t * H + h] = acc;
}

// el/er: per (node, head) dot of feat row with attn_l / attn_r
__global__ void elr_kernel(const float* __restrict__ feat, const float* __restrict__ attn_l,
                           const float* __restrict__ attn_r, float* el, float* er,
                           int n, int H, int D) {
    int gid = blockIdx.x * blockDim.x + threadIdx.x;
    if (gid >= n * H) return;
    int i = gid / H, h = gid % H;
    const float* f  = feat + (size_t)i * H * D + h * D;
    const float* wl = attn_l + h * D;
    const float* wr = attn_r + h * D;
    float sl = 0.f, sr = 0.f;
    for (int d = 0; d < D; ++d) { float fv = f[d]; sl += fv * wl[d]; sr += fv * wr[d]; }
    el[gid] = sl;
    er[gid] = sr;
}

// edge softmax per destination node (CSR gather; register max/sum)
__global__ void attn_kernel(const int* __restrict__ row_ptr, const int* __restrict__ eids,
                            const int* __restrict__ src, const int* __restrict__ etype,
                            const float* __restrict__ el, const float* __restrict__ er,
                            const float* __restrict__ ee, float* __restrict__ a,
                            const float* __restrict__ res_attn, int n, int H) {
    int gid = blockIdx.x * blockDim.x + threadIdx.x;
    if (gid >= n * H) return;
    int i = gid / H, h = gid % H;
    int s0 = row_ptr[i], s1 = row_ptr[i + 1];
    float erv = er[gid];
    float mx = -3.4e38f;
    for (int j = s0; j < s1; ++j) {
        int e = eids[j];
        float s = el[src[e] * H + h] + erv + ee[etype[e] * H + h];
        s = (s > 0.f) ? s : NEGC * s;   // leaky relu
        a[e * H + h] = s;
        mx = fmaxf(mx, s);
    }
    float z = 0.f;
    for (int j = s0; j < s1; ++j) {
        int e = eids[j];
        float ex = expf(a[e * H + h] - mx);
        a[e * H + h] = ex;
        z += ex;
    }
    float inv = (z > 0.f) ? 1.f / z : 0.f;
    for (int j = s0; j < s1; ++j) {
        int e = eids[j];
        float av = a[e * H + h] * inv;
        if (res_attn) av = av * (1.f - ALPHAC) + res_attn[e * H + h] * ALPHAC;
        a[e * H + h] = av;
    }
}

// rst[i,h,d] = sum_{e in CSR[i]} a[e,h] * feat[src[e],h,d]   (gather, no atomics)
__global__ void agg_kernel(const int* __restrict__ row_ptr, const int* __restrict__ eids,
                           const int* __restrict__ src, const float* __restrict__ a,
                           const float* __restrict__ feat, float* __restrict__ out,
                           int n, int H, int D) {
    int HD = H * D;
    int gid = blockIdx.x * blockDim.x + threadIdx.x;
    if (gid >= n * HD) return;
    int i = gid / HD, t = gid % HD;
    int h = t / D;
    int s0 = row_ptr[i], s1 = row_ptr[i + 1];
    float acc = 0.f;
    for (int j = s0; j < s1; ++j) {
        int e = eids[j];
        acc += a[e * H + h] * feat[(size_t)src[e] * HD + t];
    }
    out[gid] = acc;
}

__global__ void elu_kernel(float* x, int n) {
    int gid = blockIdx.x * blockDim.x + threadIdx.x;
    if (gid < n) {
        float v = x[gid];
        x[gid] = (v > 0.f) ? v : (expf(v) - 1.f);
    }
}

// ---------------------------------------------------------------------------
extern "C" void kernel_launch(void* const* d_in, const int* in_sizes, int n_in,
                              void* d_out, int out_size, void* d_ws, size_t ws_size,
                              hipStream_t stream)
{
    (void)in_sizes; (void)n_in; (void)out_size; (void)ws_size;
    int idx = 0;
    const float* x     = (const float*)d_in[idx++];
    const int*   src   = (const int*)d_in[idx++];
    const int*   dst   = (const int*)d_in[idx++];
    const int*   etype = (const int*)d_in[idx++];
    const float* fc_w  = (const float*)d_in[idx++];
    const float* fc_b  = (const float*)d_in[idx++];
    const float *w[3], *emb[3], *we[3], *al[3], *ar[3], *ae[3], *resw[3];
    for (int l = 0; l < 3; ++l) {
        w[l]   = (const float*)d_in[idx++];
        emb[l] = (const float*)d_in[idx++];
        we[l]  = (const float*)d_in[idx++];
        al[l]  = (const float*)d_in[idx++];
        ar[l]  = (const float*)d_in[idx++];
        ae[l]  = (const float*)d_in[idx++];
        resw[l] = (l > 0) ? (const float*)d_in[idx++] : nullptr;
    }

    // workspace carve-out
    char* ws = (char*)d_ws;
    size_t off = 0;
    auto carve = [&](size_t bytes) -> void* {
        void* p = ws + off;
        off = (off + bytes + 255) & ~(size_t)255;
        return p;
    };
    float* h0   = (float*)carve((size_t)NN * 64 * 4);
    float* hb1  = (float*)carve((size_t)NN * 256 * 4);
    float* hb2  = (float*)carve((size_t)NN * 256 * 4);
    float* feat = (float*)carve((size_t)NN * 256 * 4);
    float* el   = (float*)carve((size_t)NN * 4 * 4);
    float* er   = (float*)carve((size_t)NN * 4 * 4);
    float* aA   = (float*)carve((size_t)EE_ * 4 * 4);
    float* aB   = (float*)carve((size_t)EE_ * 4 * 4);
    float* eeT  = (float*)carve((size_t)3 * NETC * 4 * 4);
    int* row_ptr = (int*)carve((size_t)(NN + 1) * 4);
    int* counts  = (int*)carve((size_t)NN * 4);
    int* eids    = (int*)carve((size_t)EE_ * 4);
    float* out   = (float*)d_out;

    const int Hs[3] = {4, 4, 1};

    // ---- CSR by destination (deterministic) ----
    zero_i32<<<(NN + 255) / 256, 256, 0, stream>>>(counts, NN);
    count_kernel<<<(EE_ + 255) / 256, 256, 0, stream>>>(dst, counts, EE_);
    scan_kernel<<<1, 1024, 0, stream>>>(counts, row_ptr, NN);
    zero_i32<<<(NN + 255) / 256, 256, 0, stream>>>(counts, NN);
    scatter_kernel<<<(EE_ + 255) / 256, 256, 0, stream>>>(dst, row_ptr, counts, eids, EE_);
    sort_kernel<<<(NN + 255) / 256, 256, 0, stream>>>(row_ptr, eids, NN);

    // ---- per-etype attention bias per layer ----
    for (int l = 0; l < 3; ++l)
        ee_kernel<<<1, 32, 0, stream>>>(emb[l], we[l], ae[l], eeT + l * NETC * 4, Hs[l]);

    // ---- input projection: h0 = x @ fc_w + fc_b ----
    {
        dim3 g(64 / 16, (NN + 127) / 128);
        gemm_wmma<256><<<g, 256, 0, stream>>>(x, fc_w, fc_b, h0, NN, 64, 0);
    }

    // ---- layer 0 (H=4, D=64, no residual, elu, produce res_attn=aA) ----
    {
        dim3 g(256 / 16, (NN + 127) / 128);
        gemm_wmma<64><<<g, 256, 0, stream>>>(h0, w[0], nullptr, feat, NN, 256, 0);
        elr_kernel<<<(NN * 4 + 255) / 256, 256, 0, stream>>>(feat, al[0], ar[0], el, er, NN, 4, 64);
        attn_kernel<<<(NN * 4 + 255) / 256, 256, 0, stream>>>(row_ptr, eids, src, etype,
                                                              el, er, eeT, aA, nullptr, NN, 4);
        agg_kernel<<<(NN * 256 + 255) / 256, 256, 0, stream>>>(row_ptr, eids, src, aA, feat, hb1, NN, 4, 64);
        elu_kernel<<<(NN * 256 + 255) / 256, 256, 0, stream>>>(hb1, NN * 256);
    }

    // ---- layer 1 (H=4, D=64, residual GEMM, attn mixed with aA, elu) ----
    {
        dim3 g(256 / 16, (NN + 127) / 128);
        gemm_wmma<256><<<g, 256, 0, stream>>>(hb1, w[1], nullptr, feat, NN, 256, 0);
        elr_kernel<<<(NN * 4 + 255) / 256, 256, 0, stream>>>(feat, al[1], ar[1], el, er, NN, 4, 64);
        attn_kernel<<<(NN * 4 + 255) / 256, 256, 0, stream>>>(row_ptr, eids, src, etype,
                                                              el, er, eeT + NETC * 4, aB, aA, NN, 4);
        agg_kernel<<<(NN * 256 + 255) / 256, 256, 0, stream>>>(row_ptr, eids, src, aB, feat, hb2, NN, 4, 64);
        gemm_wmma<256><<<g, 256, 0, stream>>>(hb1, resw[1], nullptr, hb2, NN, 256, 1);  // += residual
        elu_kernel<<<(NN * 256 + 255) / 256, 256, 0, stream>>>(hb2, NN * 256);
    }

    // ---- layer 2 (H=1, D=16, residual, no activation; mean over 1 head = identity) ----
    {
        dim3 g(16 / 16, (NN + 127) / 128);
        gemm_wmma<256><<<g, 256, 0, stream>>>(hb2, w[2], nullptr, feat, NN, 16, 0);
        elr_kernel<<<(NN + 255) / 256, 256, 0, stream>>>(feat, al[2], ar[2], el, er, NN, 1, 16);
        attn_kernel<<<(NN + 255) / 256, 256, 0, stream>>>(row_ptr, eids, src, etype,
                                                          el, er, eeT + 2 * NETC * 4, aA, nullptr, NN, 1);
        agg_kernel<<<(NN * 16 + 255) / 256, 256, 0, stream>>>(row_ptr, eids, src, aA, feat, out, NN, 1, 16);
        gemm_wmma<256><<<g, 256, 0, stream>>>(hb2, resw[2], nullptr, out, NN, 16, 1);   // += residual
    }
}
